// ViR_23098334118050
// MI455X (gfx1250) — matrix operand order repaired and checked by reading, not compiled, not run on hardware
//
#include <hip/hip_runtime.h>
#include <hip/hip_bf16.h>

// ---------------- model constants ----------------
#define B_      4
#define C_      3
#define IMG_    256
#define P_      8
#define GRID_   32
#define NPATCH_ 1024
#define T_      1025      // tokens (cls + 1024 patches)
#define TP_     1056      // T padded to multiple of 32 (WMMA K step) : 33*32
#define EMB_    256
#define H_      8
#define S_      32        // head dim
#define F_      1024      // FF hidden
#define PD_     192       // patch dim
#define NL_     4
#define NCLS_   1000

typedef __attribute__((ext_vector_type(16))) _Float16 v16h;
typedef __attribute__((ext_vector_type(8)))  _Float16 v8h;
typedef __attribute__((ext_vector_type(8)))  float    v8f;

__device__ inline v16h v16h_zero() {
    v16h z;
#pragma unroll
    for (int i = 0; i < 16; ++i) z[i] = (_Float16)0.0f;
    return z;
}

__device__ inline v16h load_afrag(const _Float16* p) {
    // A 16x32 f16 fragment (per lane): elems 0..15 = [k .. k+16)
    v8h lo = *(const v8h*)(p);
    v8h hi = *(const v8h*)(p + 8);
    v16h a;
#pragma unroll
    for (int i = 0; i < 8; ++i) { a[i] = lo[i]; a[8 + i] = hi[i]; }
    return a;
}

// =====================================================================
// Generic batched WMMA GEMM:  C = alpha*(A @ Wt^T) + bias + Cin
//   A  : f16 [M,K] row-major (lda) ; Wt : f16 [N,K] row-major (ldb)
//   C  : f32 [M,N] (ldc) ; Cin optional f32 addend (same ldc)
// Each wave computes a 32x32 tile = 4 x v_wmma_f32_16x16x32_f16 per K-step,
// with 2 A frags / 2 B frags (full A/B reuse, interleaved accumulators so no
// WMMA->WMMA hazard NOPs).  Block = 4 waves in 2x2 -> 64x64 tile.
// GUARD=false requires M%32==0 && N%32==0: no per-lane branches in hot loop.
// =====================================================================
template<bool GUARD>
__global__ __launch_bounds__(128) void wmma_gemm_kernel(
    const _Float16* __restrict__ A,  int lda, long long sAb, long long sAh,
    const _Float16* __restrict__ Wt, int ldb, long long sBb, long long sBh,
    float* __restrict__ Cp,          int ldc, long long sCb, long long sCh,
    const float* __restrict__ Cin,            long long sIb, long long sIh,
    const float* __restrict__ bias,
    float alpha, int M, int N, int K, int Hdim)
{
    const int z    = blockIdx.z;
    const int bb   = z / Hdim;
    const int hh   = z % Hdim;
    const int lane = threadIdx.x;
    const int wave = threadIdx.y;          // 0..3, 2x2 layout
    const int m0   = blockIdx.x * 64 + (wave & 1) * 32;
    const int n0   = blockIdx.y * 64 + (wave >> 1) * 32;
    if (m0 >= M || n0 >= N) return;        // wave-uniform

    A  += bb * sAb + hh * sAh;
    Wt += bb * sBb + hh * sBh;
    Cp += bb * sCb + hh * sCh;
    if (Cin) Cin += bb * sIb + hh * sIh;

    const int r16  = lane & 15;
    const int half = lane >> 4;

    v8f acc00, acc01, acc10, acc11;
#pragma unroll
    for (int i = 0; i < 8; ++i) { acc00[i] = 0.f; acc01[i] = 0.f; acc10[i] = 0.f; acc11[i] = 0.f; }

    const _Float16* a0 = A  + (long long)(m0 + r16) * lda + half * 16;  // rows m0..m0+15
    const _Float16* a1 = a0 + (long long)16 * lda;                      // rows m0+16..m0+31
    const _Float16* b0 = Wt + (long long)(n0 + r16) * ldb + half * 16;  // cols n0..n0+15
    const _Float16* b1 = b0 + (long long)16 * ldb;                      // cols n0+16..n0+31

    bool a0ok = true, a1ok = true, b0ok = true, b1ok = true;
    if (GUARD) {
        a0ok = (m0 + r16)      < M;
        a1ok = (m0 + 16 + r16) < M;
        b0ok = (n0 + r16)      < N;
        b1ok = (n0 + 16 + r16) < N;
    }

    for (int k0 = 0; k0 < K; k0 += 32) {
        v16h A0, A1, B0, B1;
        if (GUARD) {
            A0 = a0ok ? load_afrag(a0 + k0) : v16h_zero();
            A1 = a1ok ? load_afrag(a1 + k0) : v16h_zero();
            B0 = b0ok ? *(const v16h*)(b0 + k0) : v16h_zero();
            B1 = b1ok ? *(const v16h*)(b1 + k0) : v16h_zero();
        } else {
            A0 = load_afrag(a0 + k0);
            A1 = load_afrag(a1 + k0);
            B0 = *(const v16h*)(b0 + k0);     // 32B aligned
            B1 = *(const v16h*)(b1 + k0);
            if (k0 + 32 < K) {                // wave-uniform prefetch of next K chunk
                __builtin_prefetch(a0 + k0 + 32, 0, 1);
                __builtin_prefetch(b0 + k0 + 32, 0, 1);
            }
        }
        acc00 = __builtin_amdgcn_wmma_f32_16x16x32_f16(false, A0, false, B0, (short)0, acc00, false, false);
        acc01 = __builtin_amdgcn_wmma_f32_16x16x32_f16(false, A0, false, B1, (short)0, acc01, false, false);
        acc10 = __builtin_amdgcn_wmma_f32_16x16x32_f16(false, A1, false, B0, (short)0, acc10, false, false);
        acc11 = __builtin_amdgcn_wmma_f32_16x16x32_f16(false, A1, false, B1, (short)0, acc11, false, false);
    }

    // D layout: col = lane&15 ; acc[j] -> row = half*8 + j
#pragma unroll
    for (int dm = 0; dm < 2; ++dm) {
#pragma unroll
        for (int dn = 0; dn < 2; ++dn) {
            const v8f& acc = dm ? (dn ? acc11 : acc10) : (dn ? acc01 : acc00);
            const int ccol = n0 + dn * 16 + r16;
#pragma unroll
            for (int j = 0; j < 8; ++j) {
                const int crow = m0 + dm * 16 + half * 8 + j;
                if (!GUARD || (crow < M && ccol < N)) {
                    float v = acc[j] * alpha;
                    if (bias) v += bias[ccol];
                    if (Cin)  v += Cin[(long long)crow * ldc + ccol];
                    Cp[(long long)crow * ldc + ccol] = v;
                }
            }
        }
    }
}

// ---------------- elementwise / aux kernels ----------------
__global__ void zero_f32_kernel(float* __restrict__ p, long long n) {
    long long i = (long long)blockIdx.x * blockDim.x + threadIdx.x;
    for (; i < n; i += (long long)gridDim.x * blockDim.x) p[i] = 0.0f;
}

__global__ void f32_to_f16_kernel(const float* __restrict__ s, _Float16* __restrict__ d, long long n) {
    long long i = (long long)blockIdx.x * blockDim.x + threadIdx.x;
    for (; i < n; i += (long long)gridDim.x * blockDim.x) d[i] = (_Float16)s[i];
}

// im2col: x [B,C,256,256] -> patches f16 [B,1024,192], feature order (c,kh,kw)
__global__ void im2col_kernel(const float* __restrict__ x, _Float16* __restrict__ out, long long total) {
    long long i = (long long)blockIdx.x * blockDim.x + threadIdx.x;
    for (; i < total; i += (long long)gridDim.x * blockDim.x) {
        int k = (int)(i % PD_);
        long long rest = i / PD_;
        int p = (int)(rest % NPATCH_);
        int b = (int)(rest / NPATCH_);
        int c = k / (P_ * P_), kh = (k / P_) % P_, kw = k % P_;
        int gy = p / GRID_, gx = p % GRID_;
        int row = gy * P_ + kh, col = gx * P_ + kw;
        out[i] = (_Float16)x[(((long long)b * C_ + c) * IMG_ + row) * IMG_ + col];
    }
}

// t==0 -> cls token ; t>=1 -> h += pos_emb[t-1]   (h is [B,TP_,EMB_])
__global__ void pos_cls_kernel(float* __restrict__ h, const float* __restrict__ pos,
                               const float* __restrict__ cls, long long total) {
    long long i = (long long)blockIdx.x * blockDim.x + threadIdx.x;
    for (; i < total; i += (long long)gridDim.x * blockDim.x) {
        int e = (int)(i % EMB_);
        long long rest = i / EMB_;
        int t = (int)(rest % T_);
        int b = (int)(rest / T_);
        float* dst = h + ((long long)b * TP_ + t) * EMB_ + e;
        if (t == 0) *dst = cls[e];
        else        *dst += pos[(long long)(t - 1) * EMB_ + e];
    }
}

// kqvtmp f32 [B,TP,H,96] -> k,q f16 [B,H,TP,32], vt f16 [B,H,32,TP]
__global__ void kqv_scatter_kernel(const float* __restrict__ kqv,
                                   _Float16* __restrict__ q, _Float16* __restrict__ k,
                                   _Float16* __restrict__ vt, long long total) {
    long long i = (long long)blockIdx.x * blockDim.x + threadIdx.x;
    for (; i < total; i += (long long)gridDim.x * blockDim.x) {
        int s = (int)(i % S_);
        long long rest = i / S_;
        int t  = (int)(rest % TP_); rest /= TP_;
        int hh = (int)(rest % H_);
        int b  = (int)(rest / H_);
        const float* src = kqv + (((long long)b * TP_ + t) * H_ + hh) * (3 * S_);
        long long off = (((long long)(b * H_ + hh)) * TP_ + t) * S_ + s;
        k[off] = (_Float16)src[s];
        q[off] = (_Float16)src[S_ + s];
        vt[(((long long)(b * H_ + hh)) * S_ + s) * TP_ + t] = (_Float16)src[2 * S_ + s];
    }
}

// softmax over keys j<T_ of prev row; write P f16 with zero padding (j>=T_)
__global__ __launch_bounds__(256) void softmax_kernel(const float* __restrict__ Sm,
                                                      _Float16* __restrict__ Pm) {
    __shared__ float red[256];
    const long long base = ((long long)blockIdx.y * TP_ + blockIdx.x) * TP_;
    const float* row = Sm + base;
    _Float16* prow = Pm + base;
    const int tid = threadIdx.x;

    float m = -3.4e38f;
    for (int j = tid; j < T_; j += 256) m = fmaxf(m, row[j]);
    red[tid] = m; __syncthreads();
    for (int s = 128; s > 0; s >>= 1) { if (tid < s) red[tid] = fmaxf(red[tid], red[tid + s]); __syncthreads(); }
    m = red[0]; __syncthreads();

    float sum = 0.0f;
    for (int j = tid; j < T_; j += 256) sum += __expf(row[j] - m);
    red[tid] = sum; __syncthreads();
    for (int s = 128; s > 0; s >>= 1) { if (tid < s) red[tid] += red[tid + s]; __syncthreads(); }
    const float inv = 1.0f / red[0];

    for (int j = tid; j < TP_; j += 256) {
        float v = (j < T_) ? __expf(row[j] - m) * inv : 0.0f;
        prow[j] = (_Float16)v;
    }
}

// h = LayerNorm(h + delta)*w + b ; also refresh h_f16.  One block per row (EMB_=256 threads).
__global__ __launch_bounds__(256) void add_ln_kernel(float* __restrict__ h, _Float16* __restrict__ h16,
                                                     const float* __restrict__ delta,
                                                     const float* __restrict__ w,
                                                     const float* __restrict__ bt) {
    __shared__ float red[256];
    const long long base = (long long)blockIdx.x * EMB_;
    const int e = threadIdx.x;
    float v = h[base + e] + delta[base + e];

    red[e] = v; __syncthreads();
    for (int s = 128; s > 0; s >>= 1) { if (e < s) red[e] += red[e + s]; __syncthreads(); }
    const float mu = red[0] * (1.0f / EMB_); __syncthreads();

    const float d = v - mu;
    red[e] = d * d; __syncthreads();
    for (int s = 128; s > 0; s >>= 1) { if (e < s) red[e] += red[e + s]; __syncthreads(); }
    const float var = red[0] * (1.0f / EMB_);

    const float out = d * rsqrtf(var + 1e-5f) * w[e] + bt[e];
    h[base + e]   = out;
    h16[base + e] = (_Float16)out;
}

// exact GELU (erf form), f32 in -> f16 out
__global__ void gelu_kernel(const float* __restrict__ s, _Float16* __restrict__ d, long long n) {
    long long i = (long long)blockIdx.x * blockDim.x + threadIdx.x;
    for (; i < n; i += (long long)gridDim.x * blockDim.x) {
        float x = s[i];
        d[i] = (_Float16)(0.5f * x * (1.0f + erff(x * 0.70710678118654752f)));
    }
}

// gather cls rows (t=0) into a 16x256 f16 matrix (rows 4..15 zero) for the head GEMM
__global__ void gather_cls_kernel(const _Float16* __restrict__ h16, _Float16* __restrict__ out) {
    int i = blockIdx.x * blockDim.x + threadIdx.x;
    if (i >= 16 * EMB_) return;
    int r = i / EMB_, e = i % EMB_;
    out[i] = (r < B_) ? h16[(long long)r * TP_ * EMB_ + e] : (_Float16)0.0f;
}

// =====================================================================
// host launcher
// =====================================================================
extern "C" void kernel_launch(void* const* d_in, const int* in_sizes, int n_in,
                              void* d_out, int out_size, void* d_ws, size_t ws_size,
                              hipStream_t stream) {
    (void)in_sizes; (void)n_in; (void)out_size; (void)ws_size;

    const float* x        = (const float*)d_in[0];
    const float* patch_w  = (const float*)d_in[1];
    const float* patch_b  = (const float*)d_in[2];
    const float* pos_emb  = (const float*)d_in[3];
    const float* cls_tok  = (const float*)d_in[4];
    const float* kqv_w    = (const float*)d_in[5];
    const float* proj_w   = (const float*)d_in[6];
    const float* ln1_w    = (const float*)d_in[7];
    const float* ln1_b    = (const float*)d_in[8];
    const float* ln2_w    = (const float*)d_in[9];
    const float* ln2_b    = (const float*)d_in[10];
    const float* ff1_w    = (const float*)d_in[11];
    const float* ff1_b    = (const float*)d_in[12];
    const float* ff2_w    = (const float*)d_in[13];
    const float* ff2_b    = (const float*)d_in[14];
    const float* head_w   = (const float*)d_in[15];
    const float* head_b   = (const float*)d_in[16];
    float* out            = (float*)d_out;

    // ---- workspace carve-up (256B aligned) ----
    char* base = (char*)d_ws;
    size_t off = 0;
    auto alloc = [&](size_t bytes) -> void* {
        void* p = base + off;
        off = (off + bytes + 255) & ~(size_t)255;
        return p;
    };
    const long long NTOT = (long long)B_ * TP_ * EMB_;
    const long long NSC  = (long long)B_ * H_ * TP_ * TP_;

    float*    h_f32   = (float*)   alloc(NTOT * 4);
    _Float16* h_f16   = (_Float16*)alloc(NTOT * 2);
    _Float16* patches = (_Float16*)alloc((size_t)B_ * NPATCH_ * PD_ * 2);
    _Float16* pw16    = (_Float16*)alloc((size_t)EMB_ * PD_ * 2);
    _Float16* kqvw16  = (_Float16*)alloc((size_t)NL_ * 3 * S_ * S_ * 2);
    _Float16* projw16 = (_Float16*)alloc((size_t)NL_ * EMB_ * EMB_ * 2);
    _Float16* ff1w16  = (_Float16*)alloc((size_t)NL_ * F_ * EMB_ * 2);
    _Float16* ff2w16  = (_Float16*)alloc((size_t)NL_ * EMB_ * F_ * 2);
    _Float16* headw16 = (_Float16*)alloc((size_t)NCLS_ * EMB_ * 2);
    float*    kqvtmp  = (float*)   alloc((size_t)B_ * TP_ * H_ * 3 * S_ * 4);
    _Float16* q16     = (_Float16*)alloc((size_t)B_ * H_ * TP_ * S_ * 2);
    _Float16* k16     = (_Float16*)alloc((size_t)B_ * H_ * TP_ * S_ * 2);
    _Float16* vt16    = (_Float16*)alloc((size_t)B_ * H_ * S_ * TP_ * 2);
    float*    prev    = (float*)   alloc((size_t)NSC * 4);       // residual scores (f32)
    _Float16* Pm      = (_Float16*)alloc((size_t)NSC * 2);       // softmax probs
    float*    res32   = (float*)   alloc(NTOT * 4);
    _Float16* res16   = (_Float16*)alloc(NTOT * 2);
    float*    tmp32   = (float*)   alloc(NTOT * 4);              // proj / ff2 output
    float*    ff32    = (float*)   alloc((size_t)B_ * TP_ * F_ * 4);
    _Float16* ffh16   = (_Float16*)alloc((size_t)B_ * TP_ * F_ * 2);
    _Float16* cls16   = (_Float16*)alloc((size_t)16 * EMB_ * 2);

    auto cdiv = [](long long a, long long b) { return (int)((a + b - 1) / b); };
    auto zero = [&](float* p, long long n) {
        int g = cdiv(n, 256); if (g > 8192) g = 8192;
        zero_f32_kernel<<<g, 256, 0, stream>>>(p, n);
    };
    auto cvt = [&](const float* s, _Float16* d, long long n) {
        int g = cdiv(n, 256); if (g > 8192) g = 8192;
        f32_to_f16_kernel<<<g, 256, 0, stream>>>(s, d, n);
    };
    auto gemm = [&](const _Float16* A, int lda, long long sAb, long long sAh,
                    const _Float16* Wt, int ldb, long long sBb, long long sBh,
                    float* Cp, int ldc, long long sCb, long long sCh,
                    const float* Cin, long long sIb, long long sIh,
                    const float* bias, float alpha, int M, int N, int K, int Hd, int Z,
                    bool guard) {
        dim3 blk(32, 4, 1), grd(cdiv(M, 64), cdiv(N, 64), Z);
        if (guard)
            wmma_gemm_kernel<true><<<grd, blk, 0, stream>>>(A, lda, sAb, sAh, Wt, ldb, sBb, sBh,
                                                            Cp, ldc, sCb, sCh, Cin, sIb, sIh,
                                                            bias, alpha, M, N, K, Hd);
        else
            wmma_gemm_kernel<false><<<grd, blk, 0, stream>>>(A, lda, sAb, sAh, Wt, ldb, sBb, sBh,
                                                             Cp, ldc, sCb, sCh, Cin, sIb, sIh,
                                                             bias, alpha, M, N, K, Hd);
    };

    // ---- init (padded rows & layer-0 score residual must be zero, every call) ----
    zero(h_f32, NTOT);
    zero(prev, NSC);

    // ---- weights -> f16 ----
    cvt(patch_w, pw16,   (long long)EMB_ * PD_);
    cvt(kqv_w,   kqvw16, (long long)NL_ * 3 * S_ * S_);
    cvt(proj_w,  projw16,(long long)NL_ * EMB_ * EMB_);
    cvt(ff1_w,   ff1w16, (long long)NL_ * F_ * EMB_);
    cvt(ff2_w,   ff2w16, (long long)NL_ * EMB_ * F_);
    cvt(head_w,  headw16,(long long)NCLS_ * EMB_);

    // ---- patch embedding ----
    im2col_kernel<<<8192, 256, 0, stream>>>(x, patches, (long long)B_ * NPATCH_ * PD_);
    // tok = patches @ patch_w^T + patch_b   (rows 1..1024 of h per batch)
    gemm(patches, PD_, (long long)NPATCH_ * PD_, 0,
         pw16, PD_, 0, 0,
         h_f32 + EMB_, EMB_, (long long)TP_ * EMB_, 0,
         nullptr, 0, 0, patch_b, 1.0f, NPATCH_, EMB_, PD_, 1, B_, false);
    pos_cls_kernel<<<8192, 256, 0, stream>>>(h_f32, pos_emb, cls_tok, (long long)B_ * T_ * EMB_);
    cvt(h_f32, h_f16, NTOT);

    const float scale = 0.17677669529663687f; // 1/sqrt(32)

    for (int l = 0; l < NL_; ++l) {
        // kqv: [B*TP*H, 32] @ kqv_w[l]^T -> [B*TP*H, 96]
        gemm(h_f16, S_, 0, 0,
             kqvw16 + (long long)l * 3 * S_ * S_, S_, 0, 0,
             kqvtmp, 3 * S_, 0, 0, nullptr, 0, 0, nullptr,
             1.0f, B_ * TP_ * H_, 3 * S_, S_, 1, 1, false);
        kqv_scatter_kernel<<<8192, 256, 0, stream>>>(kqvtmp, q16, k16, vt16,
                                                     (long long)B_ * H_ * TP_ * S_);
        // scores: prev = scale*(Q @ K^T) + prev   (per b,h)
        gemm(q16, S_, (long long)TP_ * S_, 0,
             k16, S_, (long long)TP_ * S_, 0,
             prev, TP_, (long long)TP_ * TP_, 0,
             prev, (long long)TP_ * TP_, 0, nullptr,
             scale, TP_, TP_, S_, 1, B_ * H_, false);
        // softmax over keys
        softmax_kernel<<<dim3(TP_, B_ * H_), 256, 0, stream>>>(prev, Pm);
        // res = P @ V   (Vt stored [s, t] so it's A@Wt^T) -> res32 [B,TP,EMB] head-sliced
        gemm(Pm, TP_, (long long)H_ * TP_ * TP_, (long long)TP_ * TP_,
             vt16, TP_, (long long)H_ * S_ * TP_, (long long)S_ * TP_,
             res32, EMB_, (long long)TP_ * EMB_, S_,
             nullptr, 0, 0, nullptr, 1.0f, TP_, S_, TP_, H_, B_ * H_, false);
        cvt(res32, res16, NTOT);
        // proj
        gemm(res16, EMB_, 0, 0,
             projw16 + (long long)l * EMB_ * EMB_, EMB_, 0, 0,
             tmp32, EMB_, 0, 0, nullptr, 0, 0, nullptr,
             1.0f, B_ * TP_, EMB_, EMB_, 1, 1, false);
        add_ln_kernel<<<B_ * TP_, 256, 0, stream>>>(h_f32, h_f16, tmp32,
                                                    ln1_w + l * EMB_, ln1_b + l * EMB_);
        // FF
        gemm(h_f16, EMB_, 0, 0,
             ff1w16 + (long long)l * F_ * EMB_, EMB_, 0, 0,
             ff32, F_, 0, 0, nullptr, 0, 0, ff1_b + l * F_,
             1.0f, B_ * TP_, F_, EMB_, 1, 1, false);
        gelu_kernel<<<8192, 256, 0, stream>>>(ff32, ffh16, (long long)B_ * TP_ * F_);
        gemm(ffh16, F_, 0, 0,
             ff2w16 + (long long)l * EMB_ * F_, F_, 0, 0,
             tmp32, EMB_, 0, 0, nullptr, 0, 0, ff2_b + l * EMB_,
             1.0f, B_ * TP_, EMB_, F_, 1, 1, false);
        add_ln_kernel<<<B_ * TP_, 256, 0, stream>>>(h_f32, h_f16, tmp32,
                                                    ln2_w + l * EMB_, ln2_b + l * EMB_);
    }

    // ---- classification head on cls tokens ----
    gather_cls_kernel<<<16, 256, 0, stream>>>(h_f16, cls16);
    gemm(cls16, EMB_, 0, 0,
         headw16, EMB_, 0, 0,
         out, NCLS_, 0, 0, nullptr, 0, 0, head_b,
         1.0f, B_, NCLS_, EMB_, 1, 1, true);
}
